// Attention_61546881351755
// MI455X (gfx1250) — matrix operand order repaired and checked by reading, not compiled, not run on hardware
//
#include <hip/hip_runtime.h>
#include <hip/hip_bf16.h>
#include <math.h>

#define EMB 1024
#define HD  128
#define NB  4
#define SEQ 4096

typedef __attribute__((ext_vector_type(16))) __bf16 v16bf;
typedef __attribute__((ext_vector_type(8)))  float  v8f;
typedef __attribute__((ext_vector_type(4)))  unsigned int u32x4;
typedef __attribute__((ext_vector_type(8)))  int i32x8;
typedef __attribute__((ext_vector_type(4)))  int i32x4;

union Frag16 { v16bf v; __bf16 e[16]; u32x4 q[2]; };
union FragC  { v8f v; float e[8]; };

static __device__ __forceinline__ v8f wmma_bf16(const Frag16& a, const Frag16& b, v8f c) {
  return __builtin_amdgcn_wmma_f32_16x16x32_bf16(false, a.v, false, b.v, (short)0, c, false, false);
}

// Low 32 bits of a generic LDS pointer == wave-relative LDS byte address (ISA 10.2).
static __device__ __forceinline__ unsigned lds_off(const void* p) {
  return (unsigned)(uintptr_t)p;
}

// Issue a 2-D TDM tile load (global -> LDS). dims/strides in elements (bf16).
static __device__ __forceinline__ void tdm_load_2d(const void* gptr, unsigned ldsaddr,
                                                   unsigned tdim0, unsigned tdim1,
                                                   unsigned tile0, unsigned tile1,
                                                   unsigned stride0) {
  unsigned long long ga = (unsigned long long)(uintptr_t)gptr;
  u32x4 g0;
  g0[0] = 1u;                                   // count=1, user descriptor
  g0[1] = ldsaddr;                              // lds_addr [63:32]
  g0[2] = (unsigned)ga;                         // global_addr [95:64]
  g0[3] = (unsigned)((ga >> 32) & 0x1FFFFFFu) | (2u << 30);  // addr hi + type=2
  i32x8 g1;
  g1[0] = 0x00010000;                           // data_size = 1 (2 bytes)
  g1[1] = (int)(tdim0 << 16);                   // tensor_dim0 [79:48] lo16
  g1[2] = (int)((tdim0 >> 16) | (tdim1 << 16)); // tensor_dim0 hi16 | tensor_dim1 lo16
  g1[3] = (int)((tdim1 >> 16) | (tile0 << 16)); // tensor_dim1 hi16 | tile_dim0
  g1[4] = (int)tile1;                           // tile_dim1 (tile_dim2 = 0)
  g1[5] = (int)stride0;                         // tensor_dim0_stride lo32
  g1[6] = 0;                                    // stride hi16 | dim1_stride lo16
  g1[7] = 0;
  i32x4 z4; z4[0] = 0; z4[1] = 0; z4[2] = 0; z4[3] = 0;
  i32x8 z8;
#pragma unroll
  for (int i = 0; i < 8; i++) z8[i] = 0;
  __builtin_amdgcn_tensor_load_to_lds(g0, g1, z4, z4, z8, 0);
}

// ---------------- Kernel 0: fp32 weights -> bf16, transposed to [HD, EMB] ----
__global__ void wt_kernel(const float* __restrict__ Wq, const float* __restrict__ Wk,
                          const float* __restrict__ Wv,
                          __bf16* __restrict__ WqT, __bf16* __restrict__ WkT,
                          __bf16* __restrict__ WvT) {
  int idx = blockIdx.x * blockDim.x + threadIdx.x;
  const int n = EMB * HD;
  if (idx >= 3 * n) return;
  int m = idx / n, r = idx - m * n;
  int k = r / HD, h = r - k * HD;
  const float* W = (m == 0) ? Wq : (m == 1) ? Wk : Wv;
  __bf16* WT     = (m == 0) ? WqT : (m == 1) ? WkT : WvT;
  WT[(size_t)h * EMB + k] = (__bf16)W[(size_t)k * HD + h];
}

// ---------------- Kernel 1: fused QKV projection (WMMA bf16) -----------------
__global__ __launch_bounds__(256)
void qkv_kernel(const float* __restrict__ x,
                const __bf16* __restrict__ WqT, const __bf16* __restrict__ WkT,
                const __bf16* __restrict__ WvT,
                __bf16* __restrict__ Q, __bf16* __restrict__ K, __bf16* __restrict__ VT) {
  __shared__ __align__(16) float ldsX[16 * 32];
  const int tb   = blockIdx.x;
  const int tid  = threadIdx.x;
  const int wave = tid >> 5;
  const int lane = tid & 31;
  const int col  = wave * 16;

  FragC cq, ck, cv;
#pragma unroll
  for (int i = 0; i < 8; i++) { cq.e[i] = 0.f; ck.e[i] = 0.f; cv.e[i] = 0.f; }

  const float* xbase = x + (size_t)tb * 16 * EMB;
  const int n  = lane & 15;
  const int g  = lane >> 4;
  const int k0 = g << 4;
  const int ks = g << 4;

  for (int kb = 0; kb < EMB; kb += 32) {
    for (int t = tid; t < 512; t += 256) {
      int r = t >> 5, c = t & 31;
      ldsX[t] = xbase[(size_t)r * EMB + kb + c];
    }
    __syncthreads();

    Frag16 a;
    {
      const float* bp = &ldsX[(lane & 15) * 32];
#pragma unroll
      for (int j = 0; j < 8; j++) {
        a.e[j]     = (__bf16)bp[k0 + j];
        a.e[8 + j] = (__bf16)bp[k0 + 8 + j];
      }
    }
    const size_t boff = (size_t)(col + n) * EMB + kb + ks;
    Frag16 bq, bk, bv;
    { const u32x4* p = (const u32x4*)(WqT + boff); bq.q[0] = p[0]; bq.q[1] = p[1]; }
    { const u32x4* p = (const u32x4*)(WkT + boff); bk.q[0] = p[0]; bk.q[1] = p[1]; }
    { const u32x4* p = (const u32x4*)(WvT + boff); bv.q[0] = p[0]; bv.q[1] = p[1]; }

    cq.v = wmma_bf16(a, bq, cq.v);
    ck.v = wmma_bf16(a, bk, ck.v);
    cv.v = wmma_bf16(a, bv, cv.v);
    __syncthreads();
  }

  const int b    = tb / (SEQ / 16);
  const int trow = (tb % (SEQ / 16)) * 16;
#pragma unroll
  for (int v = 0; v < 8; v++) {
    int m = g * 8 + v;
    size_t qoff = ((size_t)tb * 16 + m) * HD + col + n;
    Q[qoff] = (__bf16)cq.e[v];
    K[qoff] = (__bf16)ck.e[v];
    VT[(size_t)b * HD * SEQ + (size_t)(col + n) * SEQ + trow + m] = (__bf16)cv.e[v];
  }
}

// ---------------- Kernel 2: flash attention with TDM-staged K/V --------------
// block = 128 threads (4 waves) owning 64 query rows; wave w -> rows qb+16w..+15.
// K (64x128) and V^T (128x64) tiles DMA'd into double-buffered LDS by TDM.
__global__ __launch_bounds__(128)
void attn_kernel(const __bf16* __restrict__ Q, const __bf16* __restrict__ K,
                 const __bf16* __restrict__ VT, float* __restrict__ out) {
  __shared__ __align__(16) __bf16 ldsK[2][64 * HD];   // [buf][token][dim]
  __shared__ __align__(16) __bf16 ldsV[2][HD * 64];   // [buf][dim][token]
  __shared__ __align__(16) __bf16 ldsP[4][16 * 64];   // per-wave P staging

  const int wave  = threadIdx.x >> 5;
  const int lane  = threadIdx.x & 31;
  const int nqb   = SEQ / 64;
  const int b     = blockIdx.x / nqb;
  const int qb    = (blockIdx.x % nqb) * 64;
  const int qbase = qb + wave * 16;

  const __bf16* Qb = Q  + (size_t)b * SEQ * HD;
  const __bf16* Kb = K  + (size_t)b * SEQ * HD;
  const __bf16* Vb = VT + (size_t)b * HD * SEQ;

  const int n  = lane & 15;
  const int g  = lane >> 4;
  const int k0 = g << 4;
  const int ks = g << 4;

  // preload Q A-fragments (4 head-dim chunks of 32)
  Frag16 qa[4];
  {
    const size_t row = (size_t)(qbase + (lane & 15)) * HD;
#pragma unroll
    for (int hc = 0; hc < 4; hc++) {
      const __bf16* p = Qb + row + hc * 32 + k0;
      qa[hc].q[0] = ((const u32x4*)p)[0];
      qa[hc].q[1] = ((const u32x4*)(p + 8))[0];
    }
  }

  FragC o[8];
  float mmax[8], lsum[8];
#pragma unroll
  for (int v = 0; v < 8; v++) { mmax[v] = -__builtin_inff(); lsum[v] = 0.f; }
#pragma unroll
  for (int t = 0; t < 8; t++)
#pragma unroll
    for (int v = 0; v < 8; v++) o[t].e[v] = 0.f;

  // exp2-domain scale: 1/sqrt(128) * log2(e)
  const float kscale = 0.08838834764831845f * 1.4426950408889634f;

  const int nblk = qb / 64 + 1;   // block-uniform causal trip count

  if (wave == 0) {
    tdm_load_2d(Kb, lds_off(&ldsK[0][0]), HD, SEQ, HD, 64, HD);
    tdm_load_2d(Vb, lds_off(&ldsV[0][0]), SEQ, HD, 64, HD, SEQ);
  }

  for (int blk = 0; blk < nblk; blk++) {
    const int s0  = blk * 64;
    const int cur = blk & 1;
    if (wave == 0) __builtin_amdgcn_s_wait_tensorcnt(0);
    __syncthreads();   // tile `blk` visible; everyone done reading buffer cur^1
    if (wave == 0 && blk + 1 < nblk) {
      const int s1 = s0 + 64;
      tdm_load_2d(Kb + (size_t)s1 * HD, lds_off(&ldsK[cur ^ 1][0]), HD, SEQ, HD, 64, HD);
      tdm_load_2d(Vb + s1,              lds_off(&ldsV[cur ^ 1][0]), SEQ, HD, 64, HD, SEQ);
    }
    const __bf16* Kt = &ldsK[cur][0];
    const __bf16* Vt = &ldsV[cur][0];

    // ---- S = Q K^T : four 16x16 C tiles over 64 keys ----
    FragC s[4];
#pragma unroll
    for (int j = 0; j < 4; j++) {
#pragma unroll
      for (int v = 0; v < 8; v++) s[j].e[v] = 0.f;
      const __bf16* krow = Kt + (size_t)(j * 16 + n) * HD;
#pragma unroll
      for (int hc = 0; hc < 4; hc++) {
        Frag16 kf;
        const u32x4* p = (const u32x4*)(krow + hc * 32 + ks);
        kf.q[0] = p[0]; kf.q[1] = p[1];
        s[j].v = wmma_bf16(qa[hc], kf, s[j].v);
      }
    }

    // ---- online softmax over 64 keys (exp2 domain) ----
    const bool needMask = (s0 + 63 > qbase);   // wave-uniform
    float corr[8];
#pragma unroll
    for (int v = 0; v < 8; v++) {
      const int qrow = qbase + g * 8 + v;
      float e0 = s[0].e[v] * kscale, e1 = s[1].e[v] * kscale;
      float e2 = s[2].e[v] * kscale, e3 = s[3].e[v] * kscale;
      if (needMask) {
        if (s0 +      n > qrow) e0 = -__builtin_inff();
        if (s0 + 16 + n > qrow) e1 = -__builtin_inff();
        if (s0 + 32 + n > qrow) e2 = -__builtin_inff();
        if (s0 + 48 + n > qrow) e3 = -__builtin_inff();
      }
      float mx = fmaxf(fmaxf(e0, e1), fmaxf(e2, e3));
#pragma unroll
      for (int off = 1; off < 16; off <<= 1) mx = fmaxf(mx, __shfl_xor(mx, off, 32));
      const float newm = fmaxf(mmax[v], mx);
      corr[v] = exp2f(mmax[v] - newm);
      mmax[v] = newm;
      const float p0 = exp2f(e0 - newm);
      const float p1 = exp2f(e1 - newm);
      const float p2 = exp2f(e2 - newm);
      const float p3 = exp2f(e3 - newm);
      float rs = (p0 + p1) + (p2 + p3);
#pragma unroll
      for (int off = 1; off < 16; off <<= 1) rs += __shfl_xor(rs, off, 32);
      lsum[v] = lsum[v] * corr[v] + rs;
      s[0].e[v] = p0; s[1].e[v] = p1; s[2].e[v] = p2; s[3].e[v] = p3;
    }
#pragma unroll
    for (int t = 0; t < 8; t++)
#pragma unroll
      for (int v = 0; v < 8; v++) o[t].e[v] *= corr[v];

    // ---- C-layout P (16x64) -> LDS -> two A-layout fragments ----
    __bf16* P = &ldsP[wave][0];
#pragma unroll
    for (int v = 0; v < 8; v++) {
      const int m = g * 8 + v;
#pragma unroll
      for (int j = 0; j < 4; j++) P[m * 64 + j * 16 + n] = (__bf16)s[j].e[v];
    }
    Frag16 pa[2];
    {
      const __bf16* pr = P + (lane & 15) * 64;
      pa[0].q[0] = ((const u32x4*)(pr + k0))[0];
      pa[0].q[1] = ((const u32x4*)(pr + k0 + 8))[0];
      pa[1].q[0] = ((const u32x4*)(pr + 32 + k0))[0];
      pa[1].q[1] = ((const u32x4*)(pr + 32 + k0 + 8))[0];
    }
    // ---- O += P V : 8 head-column tiles x two K=32 halves ----
#pragma unroll
    for (int t = 0; t < 8; t++) {
      const __bf16* vrow = Vt + (size_t)(t * 16 + n) * 64;
#pragma unroll
      for (int h2 = 0; h2 < 2; h2++) {
        Frag16 vf;
        const u32x4* p = (const u32x4*)(vrow + h2 * 32 + ks);
        vf.q[0] = p[0]; vf.q[1] = p[1];
        o[t].v = wmma_bf16(pa[h2], vf, o[t].v);
      }
    }
  }

  // ---- normalize and store fp32 output ----
  float* ob = out + ((size_t)b * SEQ + qbase) * HD;
#pragma unroll
  for (int v = 0; v < 8; v++) {
    const int m = g * 8 + v;
    const float inv = 1.0f / lsum[v];
#pragma unroll
    for (int t = 0; t < 8; t++) ob[(size_t)m * HD + t * 16 + n] = o[t].e[v] * inv;
  }
}

extern "C" void kernel_launch(void* const* d_in, const int* in_sizes, int n_in,
                              void* d_out, int out_size, void* d_ws, size_t ws_size,
                              hipStream_t stream) {
  const float* x  = (const float*)d_in[0];
  const float* Wq = (const float*)d_in[1];
  const float* Wk = (const float*)d_in[2];
  const float* Wv = (const float*)d_in[3];
  float* out = (float*)d_out;

  char* ws = (char*)d_ws;
  const size_t wsz = (size_t)EMB * HD * sizeof(__bf16);
  const size_t qsz = (size_t)NB * SEQ * HD * sizeof(__bf16);
  __bf16* WqT = (__bf16*)(ws);
  __bf16* WkT = (__bf16*)(ws + wsz);
  __bf16* WvT = (__bf16*)(ws + 2 * wsz);
  __bf16* Q   = (__bf16*)(ws + 3 * wsz);
  __bf16* K   = (__bf16*)(ws + 3 * wsz + qsz);
  __bf16* VT  = (__bf16*)(ws + 3 * wsz + 2 * qsz);

  wt_kernel<<<(3 * EMB * HD + 255) / 256, 256, 0, stream>>>(Wq, Wk, Wv, WqT, WkT, WvT);
  qkv_kernel<<<NB * SEQ / 16, 256, 0, stream>>>(x, WqT, WkT, WvT, Q, K, VT);
  attn_kernel<<<NB * SEQ / 64, 128, 0, stream>>>(Q, K, VT, out);
}
